// Edgewise_27805618274585
// MI455X (gfx1250) — compile-verified, matrix operand order checked by reading, not run on hardware
//
#include <hip/hip_runtime.h>
#include <hip/hip_bf16.h>

// ---------------------------------------------------------------------------
// eSCN-style SO(2) edge convolution for MI455X (gfx1250).
// All large GEMMs run through v_wmma_f32_16x16x32_bf16 (bf16 in, f32 acc).
// Weights are pre-transposed to N x K so both A and B WMMA fragments are
// contiguous 16B global_load_b128 per lane: no LDS, no barriers, 4 WMMAs
// per A-fragment (16x64 strip per wave). Weights (~1.9 MB bf16) stay L2-resident.
// GEMM is templated on <HAS_SCALE, HAS_BIAS, OUT_BF16> so the K-loop has no
// branches / EXEC churn (WMMA requires EXEC == all ones).
// ---------------------------------------------------------------------------

#define NCOEF   9
#define CDIM    128
#define ROWLEN  (NCOEF * CDIM)   // 1152 floats per (edge, coeff-set)

typedef __attribute__((ext_vector_type(16))) __bf16         v16bf;
typedef __attribute__((ext_vector_type(8)))  float          v8f;
typedef __attribute__((ext_vector_type(8)))  unsigned short us8;
typedef __attribute__((ext_vector_type(16))) unsigned short us16;

__device__ __forceinline__ float bf2f(unsigned short h) {
    unsigned u = ((unsigned)h) << 16;
    return __builtin_bit_cast(float, u);
}
__device__ __forceinline__ unsigned short f2bf(float f) {
    unsigned u = __builtin_bit_cast(unsigned, f);
    u += 0x7FFFu + ((u >> 16) & 1u);      // round-to-nearest-even
    return (unsigned short)(u >> 16);
}
__device__ __forceinline__ float silu_f(float x) { return x / (1.f + __expf(-x)); }

// ---------------------------------------------------------------------------
// f32 -> bf16 conversion (x_edge)
// ---------------------------------------------------------------------------
__global__ void cvt_f32_bf16_kernel(const float* __restrict__ in,
                                    unsigned short* __restrict__ out, long n) {
    long i = (long)blockIdx.x * blockDim.x + threadIdx.x;
    if (i < n) out[i] = f2bf(in[i]);
}

// ---------------------------------------------------------------------------
// f32 (K x N) -> bf16 transposed (N x K)   (weight pre-pass, L2-resident)
// ---------------------------------------------------------------------------
__global__ void cvt_transpose_kernel(const float* __restrict__ in,
                                     unsigned short* __restrict__ out,
                                     int K, int N) {
    long i = (long)blockIdx.x * blockDim.x + threadIdx.x;
    if (i < (long)K * N) {
        int n = (int)(i / K);
        int k = (int)(i - (long)n * K);
        out[i] = f2bf(in[(long)k * N + n]);
    }
}

// ---------------------------------------------------------------------------
// Zero the output accumulator (segment_sum target)
// ---------------------------------------------------------------------------
__global__ void zero_kernel(float* __restrict__ p, long n) {
    long i = (long)blockIdx.x * blockDim.x + threadIdx.x;
    if (i < n) p[i] = 0.f;
}

// ---------------------------------------------------------------------------
// Load one bf16 WMMA fragment (16 values) from a contiguous K-run.
// ISA layout: lane half h -> elems 0..7 = K (k0+8h .. k0+8h+7),
//             elems 8..15 = K (k0+16+8h .. k0+16+8h+7).   Two b128 loads.
// ---------------------------------------------------------------------------
__device__ __forceinline__ v16bf load_frag(const unsigned short* __restrict__ p,
                                           int klo, int khi) {
    const us8 lo = *reinterpret_cast<const us8*>(p + klo);
    const us8 hi = *reinterpret_cast<const us8*>(p + khi);
    const us16 u = __builtin_shufflevector(lo, hi,
                       0,1,2,3,4,5,6,7,8,9,10,11,12,13,14,15);
    return __builtin_bit_cast(v16bf, u);
}

__device__ __forceinline__ v8f wmma_bf16(v16bf a, v16bf b, v8f c) {
    return __builtin_amdgcn_wmma_f32_16x16x32_bf16(false, a, false, b,
                                                   (short)0, c, false, false);
}

// ---------------------------------------------------------------------------
// bf16 WMMA GEMM:  C[M x N] = scale(A)[M x K] @ B[K x N] (+ bias)
//
// A (bf16), generalized row addressing:
//   row r -> edge e = r / RPE, sub j = r % RPE
//   A_row(k) = A[e*a_outer + a_base + j*a_inner + k]        (k contiguous)
// Scale (bf16, HAS_SCALE):  s(k) = S[e*s_stride + s_off + k]
// BT is bf16 pre-TRANSPOSED weights, row-major N x K.
// One wave (32 threads) per 16(M) x 64(N) strip: 4 accumulators sharing the
// A fragment -> 4 v_wmma per K-step, zero LDS, zero barriers.
// Requires M % 16 == 0, N % 64 == 0, K % 32 == 0 (true at every call site).
// ---------------------------------------------------------------------------
template <bool HAS_SCALE, bool HAS_BIAS, bool OUT_BF16>
__global__ __launch_bounds__(32)
void gemm_bf16_wmma_kernel(const unsigned short* __restrict__ A,
                           const unsigned short* __restrict__ S,
                           const unsigned short* __restrict__ BT,
                           const float* __restrict__ bias,
                           void* __restrict__ Cout,
                           int N, int K,
                           int a_outer, int a_base, int a_inner, int rpe,
                           int s_stride, int s_off) {
    const int lane = threadIdx.x;
    const int half = lane >> 4;      // K-octet select
    const int ml   = lane & 15;      // row (A) / column (B) within tile

    const int tile_m = blockIdx.x * 16;
    const int tile_n = blockIdx.y * 64;

    // Per-lane A row (and scale row) base pointers
    const int row = tile_m + ml;
    const int e   = row / rpe;
    const int j   = row - e * rpe;
    const unsigned short* __restrict__ arow =
        A + (long)e * a_outer + a_base + (long)j * a_inner;
    const unsigned short* __restrict__ srow =
        HAS_SCALE ? (S + (long)e * s_stride + s_off) : (const unsigned short*)0;

    // Per-lane B column pointers (BT is N x K row-major)
    const unsigned short* __restrict__ bt0 = BT + (long)(tile_n + 0  + ml) * K;
    const unsigned short* __restrict__ bt1 = BT + (long)(tile_n + 16 + ml) * K;
    const unsigned short* __restrict__ bt2 = BT + (long)(tile_n + 32 + ml) * K;
    const unsigned short* __restrict__ bt3 = BT + (long)(tile_n + 48 + ml) * K;

    v8f acc0 = {0.f,0.f,0.f,0.f,0.f,0.f,0.f,0.f};
    v8f acc1 = acc0, acc2 = acc0, acc3 = acc0;

#pragma unroll 2
    for (int k0 = 0; k0 < K; k0 += 32) {
        const int klo = k0 + half * 8;
        const int khi = k0 + 16 + half * 8;

        // Prefetch the next A / scale chunks (global_prefetch_b8)
        if (k0 + 32 < K) {
            __builtin_prefetch(arow + k0 + 32, 0, 1);
            if (HAS_SCALE) __builtin_prefetch(srow + k0 + 32, 0, 1);
        }

        v16bf av = load_frag(arow, klo, khi);
        if (HAS_SCALE) {                  // fused radial scaling (v_pk_mul_bf16)
            const v16bf sv = load_frag(srow, klo, khi);
#pragma unroll
            for (int i = 0; i < 16; ++i)
                av[i] = (__bf16)((float)av[i] * (float)sv[i]);
        }

        const v16bf bv0 = load_frag(bt0, klo, khi);
        const v16bf bv1 = load_frag(bt1, klo, khi);
        const v16bf bv2 = load_frag(bt2, klo, khi);
        const v16bf bv3 = load_frag(bt3, klo, khi);

        acc0 = wmma_bf16(av, bv0, acc0);
        acc1 = wmma_bf16(av, bv1, acc1);
        acc2 = wmma_bf16(av, bv2, acc2);
        acc3 = wmma_bf16(av, bv3, acc3);
    }

    // ---- epilogue: bias add + store (C/D layout: VGPR r -> M = r + 8*half)
    const int gm0 = tile_m + 8 * half;
#pragma unroll
    for (int r = 0; r < 8; ++r) {
        const long rowoff = (long)(gm0 + r) * N + tile_n + ml;
        float v0 = acc0[r], v1 = acc1[r], v2 = acc2[r], v3 = acc3[r];
        if (HAS_BIAS) {
            v0 += bias[tile_n + 0  + ml];
            v1 += bias[tile_n + 16 + ml];
            v2 += bias[tile_n + 32 + ml];
            v3 += bias[tile_n + 48 + ml];
        }
        if (OUT_BF16) {
            unsigned short* o = (unsigned short*)Cout;
            o[rowoff +  0] = f2bf(v0);
            o[rowoff + 16] = f2bf(v1);
            o[rowoff + 32] = f2bf(v2);
            o[rowoff + 48] = f2bf(v3);
        } else {
            float* o = (float*)Cout;
            o[rowoff +  0] = v0;
            o[rowoff + 16] = v1;
            o[rowoff + 32] = v2;
            o[rowoff + 48] = v3;
        }
    }
}

// ---------------------------------------------------------------------------
// LayerNorm (biased var, eps=1e-5) + SiLU, 128 channels; f32 in, bf16 out
// ---------------------------------------------------------------------------
__global__ __launch_bounds__(128)
void ln_silu_kernel(const float* __restrict__ t,
                    const float* __restrict__ g,
                    const float* __restrict__ beta,
                    unsigned short* __restrict__ h) {
    const int e = blockIdx.x, c = threadIdx.x;
    const float v = t[(long)e * 128 + c];
    __shared__ float s1[128], s2[128];
    s1[c] = v; s2[c] = v * v;
    __syncthreads();
    for (int off = 64; off > 0; off >>= 1) {
        if (c < off) { s1[c] += s1[c + off]; s2[c] += s2[c + off]; }
        __syncthreads();
    }
    const float mu  = s1[0] * (1.f / 128.f);
    const float var = s2[0] * (1.f / 128.f) - mu * mu;
    const float y   = (v - mu) * rsqrtf(var + 1e-5f) * g[c] + beta[c];
    h[(long)e * 128 + c] = f2bf(silu_f(y));
}

// ---------------------------------------------------------------------------
// msg[e] = wigner[e] (9x9) @ x[src[e]] (9x128), bf16 out
// ---------------------------------------------------------------------------
__global__ __launch_bounds__(128)
void gather_wigner_kernel(const float* __restrict__ x,
                          const int* __restrict__ edge_index,
                          const float* __restrict__ wig,
                          unsigned short* __restrict__ msg, int E) {
    const int e = blockIdx.x, c = threadIdx.x;
    const int src = edge_index[e];
    const float* xs = x + (long)src * ROWLEN;
    float xr[NCOEF];
#pragma unroll
    for (int r = 0; r < NCOEF; ++r) xr[r] = xs[r * CDIM + c];
    const float* w = wig + (long)e * (NCOEF * NCOEF);
#pragma unroll
    for (int f = 0; f < NCOEF; ++f) {
        float a = 0.f;
#pragma unroll
        for (int r = 0; r < NCOEF; ++r) a = fmaf(w[f * NCOEF + r], xr[r], a);
        msg[(long)e * ROWLEN + f * CDIM + c] = f2bf(a);
    }
}

// ---------------------------------------------------------------------------
// Complex recombination of SO(2) GEMM outputs (+ optional SiLU gating)
//   y0: [E x n0] (n0=640 conv1 incl. gating / 384 conv2)
//   y1: [2E x 512], y2: [2E x 256]  -> msg [E x 9 x 128] bf16
// ---------------------------------------------------------------------------
__global__ __launch_bounds__(128)
void combine_kernel(const float* __restrict__ y0,
                    const float* __restrict__ y1,
                    const float* __restrict__ y2,
                    unsigned short* __restrict__ msg,
                    int n0, int gate) {
    const int e = blockIdx.x, c = threadIdx.x;
    const float* y0e = y0 + (long)e * n0;
    const float* y1a = y1 + (long)(2 * e) * 512;
    const float* y1b = y1 + (long)(2 * e + 1) * 512;
    const float* y2a = y2 + (long)(2 * e) * 256;
    const float* y2b = y2 + (long)(2 * e + 1) * 256;

    float row[NCOEF];
    row[0] = y0e[c];
    row[1] = y0e[128 + c];
    row[2] = y0e[256 + c];
    row[3] = y1a[c]       - y1b[256 + c];   // r0 - i1 (lo)
    row[4] = y1a[128 + c] - y1b[384 + c];   // r0 - i1 (hi)
    row[5] = y1b[c]       + y1a[256 + c];   // r1 + i0 (lo)
    row[6] = y1b[128 + c] + y1a[384 + c];   // r1 + i0 (hi)
    row[7] = y2a[c]       - y2b[128 + c];   // r0 - i1
    row[8] = y2b[c]       + y2a[128 + c];   // r1 + i0

    if (gate) {
        const float g0 = silu_f(y0e[384 + c]);
        const float g1 = silu_f(y0e[512 + c]);
        row[0] = silu_f(row[0]);
        // GATE_IDX = [0,1,0,1,0,1,1,1] for rows 1..8
        row[1] *= g0; row[2] *= g1; row[3] *= g0; row[4] *= g1;
        row[5] *= g0; row[6] *= g1; row[7] *= g1; row[8] *= g1;
    }
#pragma unroll
    for (int f = 0; f < NCOEF; ++f)
        msg[(long)e * ROWLEN + f * CDIM + c] = f2bf(row[f]);
}

// ---------------------------------------------------------------------------
// out[dst[e]] += wigner_inv[e] (9x9) @ msg[e] (9x128)
// ---------------------------------------------------------------------------
__global__ __launch_bounds__(128)
void scatter_winv_kernel(const unsigned short* __restrict__ msg,
                         const int* __restrict__ edge_index,
                         const float* __restrict__ winv,
                         float* __restrict__ out, int E) {
    const int e = blockIdx.x, c = threadIdx.x;
    const int dst = edge_index[E + e];
    float mr[NCOEF];
#pragma unroll
    for (int r = 0; r < NCOEF; ++r)
        mr[r] = bf2f(msg[(long)e * ROWLEN + r * CDIM + c]);
    const float* w = winv + (long)e * (NCOEF * NCOEF);
#pragma unroll
    for (int f = 0; f < NCOEF; ++f) {
        float a = 0.f;
#pragma unroll
        for (int r = 0; r < NCOEF; ++r) a = fmaf(w[f * NCOEF + r], mr[r], a);
        atomicAdd(&out[(long)dst * ROWLEN + f * CDIM + c], a);
    }
}

// ---------------------------------------------------------------------------
extern "C" void kernel_launch(void* const* d_in, const int* in_sizes, int n_in,
                              void* d_out, int out_size, void* d_ws, size_t ws_size,
                              hipStream_t stream) {
    (void)n_in; (void)ws_size; (void)out_size;
    const float* x      = (const float*)d_in[0];
    const float* x_edge = (const float*)d_in[1];
    const int*   eidx   = (const int*)  d_in[3];
    const float* wig    = (const float*)d_in[4];
    const float* winv   = (const float*)d_in[5];
    const float* r1w1 = (const float*)d_in[6];
    const float* r1b1 = (const float*)d_in[7];
    const float* r1g  = (const float*)d_in[8];
    const float* r1be = (const float*)d_in[9];
    const float* r1w2 = (const float*)d_in[10];
    const float* r1b2 = (const float*)d_in[11];
    const float* c1w0 = (const float*)d_in[12];
    const float* c1b0 = (const float*)d_in[13];
    const float* c1w1 = (const float*)d_in[14];
    const float* c1w2 = (const float*)d_in[15];
    const float* r2w1 = (const float*)d_in[16];
    const float* r2b1 = (const float*)d_in[17];
    const float* r2g  = (const float*)d_in[18];
    const float* r2be = (const float*)d_in[19];
    const float* r2w2 = (const float*)d_in[20];
    const float* r2b2 = (const float*)d_in[21];
    const float* c2w0 = (const float*)d_in[22];
    const float* c2b0 = (const float*)d_in[23];
    const float* c2w1 = (const float*)d_in[24];
    const float* c2w2 = (const float*)d_in[25];

    const int  E  = in_sizes[2];                 // 50000
    const long NN = (long)in_sizes[0] / ROWLEN;  // 10000 nodes

    // ---- workspace bump allocator (256B aligned) ----
    char*  base = (char*)d_ws;
    size_t off  = 0;
    auto alloc = [&](size_t bytes) -> void* {
        void* p = base + off;
        off = (off + bytes + 255) & ~(size_t)255;
        return p;
    };
    typedef unsigned short u16;
    u16* xe_bf   = (u16*)alloc((size_t)E * 128 * 2);
    u16* wt_r1w1 = (u16*)alloc(128 * 128 * 2);   // all weights stored N x K (transposed)
    u16* wt_r1w2 = (u16*)alloc(128 * 768 * 2);
    u16* wt_r2w1 = (u16*)alloc(128 * 128 * 2);
    u16* wt_r2w2 = (u16*)alloc(128 * 768 * 2);
    u16* wt_c1w0 = (u16*)alloc(384 * 640 * 2);
    u16* wt_c1w1 = (u16*)alloc(256 * 512 * 2);
    u16* wt_c1w2 = (u16*)alloc(128 * 256 * 2);
    u16* wt_c2w0 = (u16*)alloc(384 * 384 * 2);
    u16* wt_c2w1 = (u16*)alloc(256 * 512 * 2);
    u16* wt_c2w2 = (u16*)alloc(128 * 256 * 2);
    float* t_f32 = (float*)alloc((size_t)E * 128 * 4);      // LN input (reused)
    u16* h_bf    = (u16*)alloc((size_t)E * 128 * 2);        // LN+SiLU output
    u16* rad1_bf = (u16*)alloc((size_t)E * 768 * 2);
    u16* rad2_bf = (u16*)alloc((size_t)E * 768 * 2);
    u16* msg_bf  = (u16*)alloc((size_t)E * ROWLEN * 2);     // msgA -> msgB -> msgC
    float* y0    = (float*)alloc((size_t)E * 640 * 4);      // reused (640 then 384)
    float* y1    = (float*)alloc((size_t)(2 * E) * 512 * 4);
    float* y2    = (float*)alloc((size_t)(2 * E) * 256 * 4);

    auto cvt = [&](const float* src, u16* dst, long n) {
        cvt_f32_bf16_kernel<<<dim3((unsigned)((n + 255) / 256)), 256, 0, stream>>>(src, dst, n);
    };
    auto cvtT = [&](const float* src, u16* dst, int K, int N) {
        long n = (long)K * N;
        cvt_transpose_kernel<<<dim3((unsigned)((n + 255) / 256)), 256, 0, stream>>>(src, dst, K, N);
    };

    // GEMM dispatch over the 4 template combos actually used
    enum { PLAIN_F32 = 0, PLAIN_BF16 = 1, SCALED_BIAS_F32 = 2, SCALED_F32 = 3 };
    auto gemm = [&](int kind, const u16* A, const u16* S, const u16* BT,
                    const float* bias, void* C, int M, int N, int K,
                    int a_outer, int a_base, int a_inner, int rpe, int s_off) {
        dim3 grid(M / 16, N / 64), blk(32);
        switch (kind) {
        case PLAIN_F32:
            gemm_bf16_wmma_kernel<false, true, false><<<grid, blk, 0, stream>>>(
                A, S, BT, bias, C, N, K, a_outer, a_base, a_inner, rpe, 768, s_off);
            break;
        case PLAIN_BF16:
            gemm_bf16_wmma_kernel<false, true, true><<<grid, blk, 0, stream>>>(
                A, S, BT, bias, C, N, K, a_outer, a_base, a_inner, rpe, 768, s_off);
            break;
        case SCALED_BIAS_F32:
            gemm_bf16_wmma_kernel<true, true, false><<<grid, blk, 0, stream>>>(
                A, S, BT, bias, C, N, K, a_outer, a_base, a_inner, rpe, 768, s_off);
            break;
        default:
            gemm_bf16_wmma_kernel<true, false, false><<<grid, blk, 0, stream>>>(
                A, S, BT, bias, C, N, K, a_outer, a_base, a_inner, rpe, 768, s_off);
            break;
        }
    };

    // ---- 0. zero output accumulator ----
    {
        long n = (long)NN * ROWLEN;
        zero_kernel<<<dim3((unsigned)((n + 255) / 256)), 256, 0, stream>>>((float*)d_out, n);
    }
    // ---- 1. convert x_edge; transpose+convert weights (L2-resident) ----
    cvt(x_edge, xe_bf, (long)E * 128);
    cvtT(r1w1, wt_r1w1, 128, 128);  cvtT(r1w2, wt_r1w2, 128, 768);
    cvtT(r2w1, wt_r2w1, 128, 128);  cvtT(r2w2, wt_r2w2, 128, 768);
    cvtT(c1w0, wt_c1w0, 384, 640);  cvtT(c1w1, wt_c1w1, 256, 512);
    cvtT(c1w2, wt_c1w2, 128, 256);  cvtT(c2w0, wt_c2w0, 384, 384);
    cvtT(c2w1, wt_c2w1, 256, 512);  cvtT(c2w2, wt_c2w2, 128, 256);

    // ---- 2. radial MLP #1:  rad1 = (silu(LN(xe@w1+b1)))@w2 + b2  (bf16) ----
    gemm(PLAIN_F32,  xe_bf, nullptr, wt_r1w1, r1b1, t_f32,   E, 128, 128, 128, 0, 0, 1, 0);
    ln_silu_kernel<<<E, 128, 0, stream>>>(t_f32, r1g, r1be, h_bf);
    gemm(PLAIN_BF16, h_bf,  nullptr, wt_r1w2, r1b2, rad1_bf, E, 768, 128, 128, 0, 0, 1, 0);

    // ---- 3. radial MLP #2 ----
    gemm(PLAIN_F32,  xe_bf, nullptr, wt_r2w1, r2b1, t_f32,   E, 128, 128, 128, 0, 0, 1, 0);
    ln_silu_kernel<<<E, 128, 0, stream>>>(t_f32, r2g, r2be, h_bf);
    gemm(PLAIN_BF16, h_bf,  nullptr, wt_r2w2, r2b2, rad2_bf, E, 768, 128, 128, 0, 0, 1, 0);

    // ---- 4. gather + Wigner rotation:  msgA = wigner @ x[src]  (bf16) ----
    gather_wigner_kernel<<<E, 128, 0, stream>>>(x, eidx, wig, msg_bf, E);

    // ---- 5. SO(2) conv #1 (radial-scaled WMMA GEMMs) ----
    gemm(SCALED_BIAS_F32, msg_bf, rad1_bf, wt_c1w0, c1b0,   y0, E,   640, 384, ROWLEN, 0,   0,   1, 0);
    gemm(SCALED_F32,      msg_bf, rad1_bf, wt_c1w1, nullptr, y1, 2*E, 512, 256, ROWLEN, 384, 256, 2, 384);
    gemm(SCALED_F32,      msg_bf, rad1_bf, wt_c1w2, nullptr, y2, 2*E, 256, 128, ROWLEN, 896, 128, 2, 640);
    combine_kernel<<<E, 128, 0, stream>>>(y0, y1, y2, msg_bf, 640, 1);

    // ---- 6. SO(2) conv #2 ----
    gemm(SCALED_BIAS_F32, msg_bf, rad2_bf, wt_c2w0, c2b0,   y0, E,   384, 384, ROWLEN, 0,   0,   1, 0);
    gemm(SCALED_F32,      msg_bf, rad2_bf, wt_c2w1, nullptr, y1, 2*E, 512, 256, ROWLEN, 384, 256, 2, 384);
    gemm(SCALED_F32,      msg_bf, rad2_bf, wt_c2w2, nullptr, y2, 2*E, 256, 128, ROWLEN, 896, 128, 2, 640);
    combine_kernel<<<E, 128, 0, stream>>>(y0, y1, y2, msg_bf, 384, 0);

    // ---- 7. inverse Wigner + segment-sum scatter ----
    scatter_winv_kernel<<<E, 128, 0, stream>>>(msg_bf, eidx, winv, (float*)d_out, E);
}
